// InfNet_24876450578529
// MI455X (gfx1250) — compile-verified, matrix operand order checked by reading, not run on hardware
//
#include <hip/hip_runtime.h>
#include <hip/hip_bf16.h>

// ---------------------------------------------------------------------------
// GCN (2-layer) + Weibull head for MI455X (gfx1250, wave32, WMMA, TDM).
//
//   deg/dinv -> bf16 conversions -> WMMA GEMM1 (TDM-staged A tile in LDS)
//   -> float4 edge scatter (f32 atomics) -> softplus -> WMMA GEMM2 (TDM)
//   -> float4 edge scatter -> softplus -> head.
//
// GEMMs: v_wmma_f32_16x16x32_bf16, A tile (16 x K bf16) staged into LDS by
// one tensor_load_to_lds per block (shared by 4 waves), B via b128 global.
// Aggregation/activations stay fp32. h matrices (51.2MB / 26.4MB) fit in the
// 192MB L2, so the edge gather/scatter is L2-resident.
//
// Workspace layout (bytes), total ~158.4 MB:
//   [0       ] dinv           50000 f32
//   [256 KiB ] W0t  bf16      256x512          (col-major, contiguous K)
//   [512 KiB ] W1t  bf16      144x256          (col-major, zero-padded cols)
//   [1 MiB   ] Xbf  bf16      50000x512  -> reused as H1bf (50000x256)
//   [+50 MiB ] H1pre f32      50000x256  -> reused as H2pre (50000x144)
//   [+50 MiB ] H1agg f32      50000x256  -> reused as H2agg (50000x132)
// ---------------------------------------------------------------------------

typedef __attribute__((ext_vector_type(16))) __bf16        v16bf;
typedef __attribute__((ext_vector_type(8)))  float         v8f;
typedef __attribute__((ext_vector_type(4)))  unsigned int  v4u;
typedef __attribute__((ext_vector_type(8)))  int           v8i;
typedef __attribute__((ext_vector_type(4)))  int           v4i;

static __device__ __forceinline__ unsigned short f2bf(float f) {
  // round-to-nearest-even fp32 -> bf16
  unsigned u = __float_as_uint(f);
  unsigned r = (u + 0x7fffu + ((u >> 16) & 1u)) >> 16;
  return (unsigned short)r;
}

// ---------------- small utility kernels ----------------

__global__ void fill_f32(float* __restrict__ p, float v, long long n) {
  long long i = (long long)blockIdx.x * blockDim.x + threadIdx.x;
  if (i < n) p[i] = v;
}

__global__ void edge_deg(const int* __restrict__ dst, float* __restrict__ deg, int E) {
  int e = blockIdx.x * blockDim.x + threadIdx.x;
  if (e < E) atomicAdd(&deg[dst[e]], 1.0f);
}

__global__ void rsqrt_inplace(float* __restrict__ p, int n) {
  int i = blockIdx.x * blockDim.x + threadIdx.x;
  if (i < n) p[i] = rsqrtf(p[i]);
}

__global__ void f32_to_bf16(const float* __restrict__ in, unsigned short* __restrict__ out,
                            long long n) {
  long long i = (long long)blockIdx.x * blockDim.x + threadIdx.x;
  if (i < n) out[i] = f2bf(in[i]);
}

// Wt[n*K + k] = (n < N) ? bf16(W[k*N + n]) : 0   (transpose + pad columns)
__global__ void transpose_to_bf16(const float* __restrict__ W, unsigned short* __restrict__ Wt,
                                  int K, int N, int Npad) {
  int idx = blockIdx.x * blockDim.x + threadIdx.x;
  if (idx >= Npad * K) return;
  int n = idx / K;
  int k = idx - n * K;
  Wt[(size_t)n * K + k] = (n < N) ? f2bf(W[(size_t)k * N + n]) : (unsigned short)0;
}

// ---------------- WMMA bf16 GEMM with TDM-staged A tile --------------------
// C(M x Npad) = A(M x K) * Bt(Npad x K)^T.  One 16x16 tile per wave, 4 waves
// per 128-thread block.  The block's 16-row A tile (16*K bf16) is DMA'd into
// LDS once via TENSOR_LOAD_TO_LDS (D# per CDNA5 ISA 08_async_tensor §8),
// then all waves build A fragments from LDS; B comes straight from global.
//
// Fragment packing per CDNA5 ISA 7.12.2 (16-bit A 16x32 layout):
//   lane L: row/col = L&15, khalf = (L&16)?8:0
//   vgpr pair j in 0..3: K = khalf + 2j,+1 ; j in 4..7: K = 16 + khalf + 2(j-4),+1
// C/D: lane L holds col = L&15; vgpr r holds row = r + ((L&16)?8:0).
#define TDM_MAXK 512

__global__ void gemm_bf16_wmma_tdm(const unsigned short* __restrict__ A,
                                   const unsigned short* __restrict__ Bt,
                                   float* __restrict__ C,
                                   int Mtot, int K, int Npad, int ldc) {
  __shared__ unsigned short Als[16 * TDM_MAXK];

  const int lane  = threadIdx.x & 31;
  const int wave  = threadIdx.x >> 5;
  const int mTile = blockIdx.x;

  if (wave == 0) {
    // ---- Tensor DMA: 2-D tile, 16 rows x K cols of 2-byte elements ----
    unsigned long long ga = (unsigned long long)(uintptr_t)(A + (size_t)mTile * 16 * K);
    unsigned ldsOff = (unsigned)(uintptr_t)(&Als[0]);   // low 32 bits = LDS offset

    v4u g0;
    g0[0] = 1u;                                          // count=1, user desc, no gather
    g0[1] = ldsOff;                                      // lds_addr
    g0[2] = (unsigned)(ga & 0xffffffffu);                // global_addr[31:0]
    g0[3] = (unsigned)((ga >> 32) & 0x01ffffffu)         // global_addr[56:32]
            | (2u << 30);                                // type = 2 ("image")

    v8i g1;
    g1[0] = (int)(1u << 16);                             // data_size=1 (2B); no mcast/pad
    g1[1] = (int)(((unsigned)K & 0xffffu) << 16);        // tensor_dim0[15:0]
    g1[2] = (int)((((unsigned)K >> 16) & 0xffffu)        // tensor_dim0[31:16]
            | (((unsigned)Mtot & 0xffffu) << 16));       // tensor_dim1[15:0]
    g1[3] = (int)((((unsigned)Mtot >> 16) & 0xffffu)     // tensor_dim1[31:16]
            | (((unsigned)K & 0xffffu) << 16));          // tile_dim0 = K
    g1[4] = 16;                                          // tile_dim1 = 16 rows
    g1[5] = K;                                           // tensor_dim0_stride[31:0]
    g1[6] = 0;
    g1[7] = 0;

    v4i gz4 = {0, 0, 0, 0};                              // groups 2/3 unused (2-D)
    v8i gz8 = {0, 0, 0, 0, 0, 0, 0, 0};
    __builtin_amdgcn_tensor_load_to_lds(g0, g1, gz4, gz4, gz8, 0);
    __builtin_amdgcn_s_wait_tensorcnt(0);
  }
  __syncthreads();

  const int nTiles = Npad >> 4;
  int nTile = blockIdx.y * (blockDim.x >> 5) + wave;
  const bool valid = (nTile < nTiles);
  if (!valid) nTile = 0;                                 // clamp; discard store later

  const int rc    = lane & 15;
  const int khalf = (lane & 16) ? 8 : 0;

  const unsigned short* Arow = Als + (size_t)rc * K;                 // LDS
  const unsigned short* Bcol = Bt + (size_t)(nTile * 16 + rc) * K;   // global

  v8f acc = {};
  for (int k0 = 0; k0 < K; k0 += 32) {
    v16bf a, b;
#pragma unroll
    for (int j = 0; j < 8; ++j) {
      const int kk = k0 + khalf + ((j < 4) ? (2 * j) : (8 + 2 * j)); // even -> dword aligned
      const unsigned pa = *(const unsigned*)(Arow + kk);             // ds_load
      const unsigned pb = *(const unsigned*)(Bcol + kk);             // global_load
      a[2 * j]     = __builtin_bit_cast(__bf16, (unsigned short)(pa & 0xffffu));
      a[2 * j + 1] = __builtin_bit_cast(__bf16, (unsigned short)(pa >> 16));
      b[2 * j]     = __builtin_bit_cast(__bf16, (unsigned short)(pb & 0xffffu));
      b[2 * j + 1] = __builtin_bit_cast(__bf16, (unsigned short)(pb >> 16));
    }
    acc = __builtin_amdgcn_wmma_f32_16x16x32_bf16(
        /*neg_a=*/false, a, /*neg_b=*/false, b,
        /*c_mod=*/(short)0, acc, /*reuse_a=*/false, /*reuse_b=*/false);
  }

  if (valid) {
    const int m0  = mTile * 16 + ((lane & 16) ? 8 : 0);
    const int col = nTile * 16 + rc;
#pragma unroll
    for (int r = 0; r < 8; ++r)
      C[(size_t)(m0 + r) * ldc + col] = acc[r];
  }
}

// ---------------- message passing ----------------

// self-loop contribution: Hagg[i][f] = Hpre[i][f] * dinv[i]^2
__global__ void self_loop_init(const float* __restrict__ Hpre, const float* __restrict__ dinv,
                               float* __restrict__ Hagg, int n, int F, int ld_in, int ld_out) {
  long long idx = (long long)blockIdx.x * blockDim.x + threadIdx.x;
  if (idx >= (long long)n * F) return;
  int i = (int)(idx / F);
  int f = (int)(idx - (long long)i * F);
  float d = dinv[i];
  Hagg[(size_t)i * ld_out + f] = Hpre[(size_t)i * ld_in + f] * d * d;
}

// per-(edge, 4-feature) scatter-add: float4 gather + 4 f32 atomics.
// FV = F/4 (compile-time so div/mod are cheap).
template <int FV>
__global__ void edge_agg_vec4(const int* __restrict__ src, const int* __restrict__ dst,
                              const float* __restrict__ dinv,
                              const float* __restrict__ Hpre, float* __restrict__ Hagg,
                              int E, int ld_in, int ld_out) {
  long long idx = (long long)blockIdx.x * blockDim.x + threadIdx.x;
  if (idx >= (long long)E * FV) return;
  int e  = (int)(idx / FV);
  int f4 = (int)(idx - (long long)e * FV) * 4;
  int s = src[e], d = dst[e];
  float nrm = dinv[s] * dinv[d];
  const float4 v = *(const float4*)(Hpre + (size_t)s * ld_in + f4);  // b128 gather
  float* base = Hagg + (size_t)d * ld_out + f4;
  atomicAdd(base + 0, v.x * nrm);
  atomicAdd(base + 1, v.y * nrm);
  atomicAdd(base + 2, v.z * nrm);
  atomicAdd(base + 3, v.w * nrm);
}

__global__ void softplus_inplace(float* __restrict__ p, long long n) {
  long long i = (long long)blockIdx.x * blockDim.x + threadIdx.x;
  if (i >= n) return;
  float x = p[i];
  p[i] = (x > 20.0f) ? x : log1pf(expf(x));
}

// head: lbd = h[:, :128]; kappa = h[:,128] + 0.1; z = lbd * exp(lgamma(1 + 1/kappa))
// h2 has row stride 132 (float4-aligned padded layout).
// out = [ z (N x 128) | lbd (N x 128) | kappa (N x 1) ]
__global__ void head_kernel(const float* __restrict__ h2, float* __restrict__ out, int n) {
  long long idx = (long long)blockIdx.x * blockDim.x + threadIdx.x;
  if (idx >= (long long)n * 128) return;
  int i = (int)(idx >> 7);
  int f = (int)(idx & 127);
  float kappa = h2[(size_t)i * 132 + 128] + 0.1f;
  float g     = expf(lgammaf(1.0f + 1.0f / kappa));
  float lbd   = h2[(size_t)i * 132 + f];
  out[idx]                      = lbd * g;                // z
  out[(long long)n * 128 + idx] = lbd;                    // lbd
  if (f == 0) out[(long long)n * 256 + i] = kappa;        // kappa
}

// ---------------- launcher ----------------

static inline long long cdivll(long long a, long long b) { return (a + b - 1) / b; }

extern "C" void kernel_launch(void* const* d_in, const int* in_sizes, int n_in,
                              void* d_out, int out_size, void* d_ws, size_t ws_size,
                              hipStream_t stream) {
  const float* x   = (const float*)d_in[0];
  const int*  eidx = (const int*)d_in[1];
  const float* W0  = (const float*)d_in[2];
  const float* W1  = (const float*)d_in[3];

  const int N   = in_sizes[0] / 512;   // 50000 nodes (multiple of 16)
  const int IN  = 512, H1 = 256, H2 = 129, H2P = 144, H2A = 132;
  const int E   = in_sizes[1] / 2;     // 1,600,000 edges
  const int* src = eidx;
  const int* dst = eidx + E;

  char* ws = (char*)d_ws;
  float*          dinv  = (float*)(ws);
  unsigned short* W0t   = (unsigned short*)(ws + (256u << 10));
  unsigned short* W1t   = (unsigned short*)(ws + (512u << 10));
  unsigned short* Xbf   = (unsigned short*)(ws + (1u << 20));          // reused as H1bf
  float*          bufB  = (float*)(ws + (1u << 20) + 52428800u);       // H1pre -> H2pre
  float*          bufC  = (float*)(ws + (1u << 20) + 2u * 52428800u);  // H1agg -> H2agg
  unsigned short* H1bf  = Xbf;
  float* H1pre = bufB, *H1agg = bufC, *H2pre = bufB, *H2agg = bufC;

  const int T = 256;
  float* out = (float*)d_out;

  // --- degrees / normalization ---
  fill_f32<<<(unsigned)cdivll(N, T), T, 0, stream>>>(dinv, 1.0f, N);   // self loop
  edge_deg<<<(unsigned)cdivll(E, T), T, 0, stream>>>(dst, dinv, E);
  rsqrt_inplace<<<(unsigned)cdivll(N, T), T, 0, stream>>>(dinv, N);

  // --- bf16 conversions ---
  f32_to_bf16<<<(unsigned)cdivll((long long)N * IN, T), T, 0, stream>>>(x, Xbf, (long long)N * IN);
  transpose_to_bf16<<<(unsigned)cdivll((long long)H1 * IN, T), T, 0, stream>>>(W0, W0t, IN, H1, H1);
  transpose_to_bf16<<<(unsigned)cdivll((long long)H2P * H1, T), T, 0, stream>>>(W1, W1t, H1, H2, H2P);

  // --- layer 1: GEMM (WMMA + TDM) + message passing + softplus ---
  {
    dim3 g(N / 16, H1 / 64);                 // 4 waves * 16 cols = 64 cols per block
    gemm_bf16_wmma_tdm<<<g, 128, 0, stream>>>(Xbf, W0t, H1pre, N, IN, H1, H1);
  }
  self_loop_init<<<(unsigned)cdivll((long long)N * H1, T), T, 0, stream>>>(
      H1pre, dinv, H1agg, N, H1, H1, H1);
  edge_agg_vec4<64><<<(unsigned)cdivll((long long)E * 64, T), T, 0, stream>>>(
      src, dst, dinv, H1pre, H1agg, E, H1, H1);
  softplus_inplace<<<(unsigned)cdivll((long long)N * H1, T), T, 0, stream>>>(
      H1agg, (long long)N * H1);

  // --- layer 2: convert, GEMM (WMMA + TDM), message passing, softplus ---
  f32_to_bf16<<<(unsigned)cdivll((long long)N * H1, T), T, 0, stream>>>(
      H1agg, H1bf, (long long)N * H1);
  {
    dim3 g(N / 16, (unsigned)cdivll(H2P / 16, 4)); // 9 tiles over 3 blocks of 4 waves
    gemm_bf16_wmma_tdm<<<g, 128, 0, stream>>>(H1bf, W1t, H2pre, N, H1, H2P, H2P);
  }
  // aggregate 132 cols (128 lbd + kappa + 3 zero-padded) for float4 alignment
  self_loop_init<<<(unsigned)cdivll((long long)N * H2A, T), T, 0, stream>>>(
      H2pre, dinv, H2agg, N, H2A, H2P, H2A);
  edge_agg_vec4<33><<<(unsigned)cdivll((long long)E * 33, T), T, 0, stream>>>(
      src, dst, dinv, H2pre, H2agg, E, H2P, H2A);
  softplus_inplace<<<(unsigned)cdivll((long long)N * H2A, T), T, 0, stream>>>(
      H2agg, (long long)N * H2A);

  // --- head ---
  head_kernel<<<(unsigned)cdivll((long long)N * 128, T), T, 0, stream>>>(H2agg, out, N);
}
